// HTA_5677946766007
// MI455X (gfx1250) — compile-verified, matrix operand order checked by reading, not run on hardware
//
#include <hip/hip_runtime.h>
#include <hip/hip_bf16.h>

typedef unsigned short u16;
typedef unsigned int   u32;
typedef __attribute__((ext_vector_type(16))) __bf16 v16bf;
typedef __attribute__((ext_vector_type(8)))  float  v8f;
typedef __attribute__((ext_vector_type(8)))  u32    v8u;

#define CC    128
#define NHD   4
#define HD    32
#define HH    112
#define WW2   112
#define NTOK  (HH*WW2)      // 12544
#define BATCH 8
#define BNTOK (BATCH*NTOK)  // 100352
#define HID   512
#define MKV   196
#define MKVP  224

// ---------------- helpers ----------------
__device__ __forceinline__ u16 f2bf(float f){
  u32 u = __builtin_bit_cast(u32, f);
  u32 r = u + 0x7FFFu + ((u >> 16) & 1u);
  return (u16)(r >> 16);
}
__device__ __forceinline__ v8f vzero(){
  v8f v;
#pragma unroll
  for (int i=0;i<8;++i) v[i]=0.f;
  return v;
}
// Load one 16-bit WMMA operand fragment (A layout; B uses identical per-lane K
// pattern with N replacing M): lane<16 -> K {kb..kb+7, kb+16..kb+23},
// lane>=16 -> K {kb+8..kb+15, kb+24..kb+31}.
__device__ __forceinline__ v16bf load_frag(const u16* row, int kbase, int lane){
  int ksel = (lane >> 4) << 3;
  const uint4* p0 = (const uint4*)(row + kbase + ksel);
  const uint4* p1 = (const uint4*)(row + kbase + ksel + 16);
  uint4 a = *p0, b = *p1;
  v8u u; u[0]=a.x;u[1]=a.y;u[2]=a.z;u[3]=a.w;u[4]=b.x;u[5]=b.y;u[6]=b.z;u[7]=b.w;
  return __builtin_bit_cast(v16bf, u);
}
__device__ __forceinline__ void wmma(v8f& c, v16bf a, v16bf b){
  c = __builtin_amdgcn_wmma_f32_16x16x32_bf16(false, a, false, b, (short)0, c, false, false);
}
__device__ __forceinline__ float gelu_f(float x){
  return 0.5f * x * (1.f + erff(x * 0.70710678118654752f));
}

// ---- CDNA5 async copy to LDS (ASYNCcnt-tracked), 16B per lane ----
__device__ __forceinline__ void async_cp16(const u16* g, u16* l){
  unsigned lds = (unsigned)(uintptr_t)l;   // low 32 bits of generic LDS addr == LDS offset
  asm volatile("global_load_async_to_lds_b128 %0, %1, off"
               :: "v"(lds), "v"(g) : "memory");
}
#if __has_builtin(__builtin_amdgcn_s_wait_asynccnt)
#define WAIT_ASYNC(n) __builtin_amdgcn_s_wait_asynccnt(n)
#else
#define WAIT_ASYNC(n) asm volatile("s_wait_asynccnt %0" :: "i"(n) : "memory")
#endif

// ---------------- LayerNorm (fp32 in -> bf16 out), C=128, one wave per token
__global__ __launch_bounds__(256) void ln_k(const float* __restrict__ x, const float* __restrict__ g,
                                            const float* __restrict__ bta, u16* __restrict__ y, int T){
  int wave = threadIdx.x >> 5, lane = threadIdx.x & 31;
  int t = blockIdx.x * 8 + wave;
  if (t >= T) return;
  const float* p = x + (size_t)t * CC;
  float v[4]; float s=0.f, s2=0.f;
#pragma unroll
  for (int j=0;j<4;++j){ v[j] = p[lane + 32*j]; s += v[j]; s2 += v[j]*v[j]; }
#pragma unroll
  for (int m=16;m>=1;m>>=1){ s += __shfl_xor(s, m, 32); s2 += __shfl_xor(s2, m, 32); }
  float mean = s * (1.f/CC);
  float var  = s2 * (1.f/CC) - mean*mean;
  float inv  = rsqrtf(var + 1e-5f);
  u16* yp = y + (size_t)t * CC;
#pragma unroll
  for (int j=0;j<4;++j){ int c = lane + 32*j; yp[c] = f2bf((v[j]-mean)*inv*g[c] + bta[c]); }
}

// ---------------- weight transpose fp32 [K,N] -> bf16 [N,K]
__global__ __launch_bounds__(256) void wtr_k(const float* __restrict__ w, u16* __restrict__ wt, int K, int N){
  int i = blockIdx.x*256 + threadIdx.x;
  if (i >= K*N) return;
  int k = i / N, n = i % N;
  wt[(size_t)n*K + k] = f2bf(w[i]);
}

// ---------------- bf16 WMMA GEMM with double-buffered async LDS staging
// out[M,N] = A[M,K] * Wt[N,K]^T + bias (+res). Block tile 128x64, K chunk 64.
template<bool RES, bool OUTB>
__global__ __launch_bounds__(256) void gemm_k(const u16* __restrict__ A, const u16* __restrict__ Wt,
    const float* __restrict__ bias, const float* __restrict__ res,
    float* __restrict__ outF, u16* __restrict__ outB, int M, int N, int K){
  __shared__ __align__(16) u16 Ab[2][128*64];
  __shared__ __align__(16) u16 Bb[2][64*64];
  int tid = threadIdx.x;
  int wave = tid >> 5, lane = tid & 31;
  int l15 = lane & 15, hi = lane >> 4;
  int row0 = blockIdx.x * 128;
  int col0 = blockIdx.y * 64;

  // 1536 x 16B slots per K-chunk: 1024 for A (128 rows x 8), 512 for B (64 x 8).
  // 6 slots per thread; slot->matrix split is wave-uniform (1024 == 4*256).
  auto issue = [&](int buf, int k0){
#pragma unroll
    for (int j=0;j<6;++j){
      int s = tid + 256*j;
      if (s < 1024){
        int r = s >> 3, ch = s & 7;
        int gr = row0 + r; if (gr > M-1) gr = M-1;
        async_cp16(A + (size_t)gr*K + k0 + ch*8, &Ab[buf][r*64 + ch*8]);
      } else {
        int s2 = s - 1024;
        int n = s2 >> 3, ch = s2 & 7;
        async_cp16(Wt + (size_t)(col0+n)*K + k0 + ch*8, &Bb[buf][n*64 + ch*8]);
      }
    }
  };

  int mrow = (wave & 3) * 32;   // wave owns 32x32 output
  int ncol = (wave >> 2) * 32;
  v8f c00=vzero(), c01=vzero(), c10=vzero(), c11=vzero();
  int KC = K >> 6;

  issue(0, 0);
  for (int kc=0;kc<KC;++kc){
    int buf = kc & 1;
    if (kc+1 < KC){ issue(buf^1, (kc+1)<<6); WAIT_ASYNC(6); }
    else          { WAIT_ASYNC(0); }
    __syncthreads();
#pragma unroll
    for (int ks=0;ks<64;ks+=32){
      v16bf a0 = load_frag(&Ab[buf][(mrow + l15)*64],      ks, lane);
      v16bf a1 = load_frag(&Ab[buf][(mrow + 16 + l15)*64], ks, lane);
      v16bf b0 = load_frag(&Bb[buf][(ncol + l15)*64],      ks, lane);
      v16bf b1 = load_frag(&Bb[buf][(ncol + 16 + l15)*64], ks, lane);
      wmma(c00, a0, b0); wmma(c01, a0, b1);
      wmma(c10, a1, b0); wmma(c11, a1, b1);
    }
    __syncthreads();
  }

  int cc0 = col0 + ncol + l15, cc1 = cc0 + 16;
  float bi0 = bias[cc0], bi1 = bias[cc1];
#pragma unroll
  for (int r=0;r<8;++r){
    int rr0 = row0 + mrow + r + 8*hi;
    int rr1 = rr0 + 16;
    if (rr0 < M){
      size_t o0 = (size_t)rr0 * N + cc0, o1 = (size_t)rr0 * N + cc1;
      float v0 = c00[r] + bi0, v1 = c01[r] + bi1;
      if (RES){ v0 += res[o0]; v1 += res[o1]; }
      if (OUTB){ outB[o0] = f2bf(v0); outB[o1] = f2bf(v1); }
      else     { outF[o0] = v0;       outF[o1] = v1; }
    }
    if (rr1 < M){
      size_t o0 = (size_t)rr1 * N + cc0, o1 = (size_t)rr1 * N + cc1;
      float v0 = c10[r] + bi0, v1 = c11[r] + bi1;
      if (RES){ v0 += res[o0]; v1 += res[o1]; }
      if (OUTB){ outB[o0] = f2bf(v0); outB[o1] = f2bf(v1); }
      else     { outF[o0] = v0;       outF[o1] = v1; }
    }
  }
}

// ---------------- im2col for 8x8/stride-8 conv: src bf16 [B,12544,128] -> dst bf16 [1568,8192]
__global__ __launch_bounds__(256) void im2col_k(const u16* __restrict__ src, u16* __restrict__ dst){
  int i = blockIdx.x*256 + threadIdx.x;
  if (i >= 1568*8192) return;
  int kidx = i & 8191;
  int row  = i >> 13;
  int c  = kidx & 127;
  int kx = (kidx >> 7) & 7;
  int ky = (kidx >> 10) & 7;
  int b = row / 196; int rem = row % 196;
  int oy = rem / 14, ox = rem % 14;
  int gy = oy*8 + ky, gx = ox*8 + kx;
  dst[i] = src[((size_t)(b*NTOK + gy*WW2 + gx))*CC + c];
}

// ---------------- depthwise 3x3 SAME + GELU: fp32 [B,112,112,512] -> bf16
__global__ __launch_bounds__(256) void dw_k(const float* __restrict__ hsrc, const float* __restrict__ w,
                                            const float* __restrict__ bias, u16* __restrict__ out, int total){
  int i = blockIdx.x*256 + threadIdx.x;
  if (i >= total) return;
  int c = i % HID;
  int t = i / HID;
  int x = t % WW2; int y = (t / WW2) % HH; int b = t / (WW2*HH);
  float acc = bias[c];
#pragma unroll
  for (int dy=-1;dy<=1;++dy){
#pragma unroll
    for (int dx=-1;dx<=1;++dx){
      int yy=y+dy, xx=x+dx;
      if (yy>=0 && yy<HH && xx>=0 && xx<WW2)
        acc += hsrc[(((size_t)b*HH + yy)*WW2 + xx)*HID + c] * w[((dy+1)*3 + (dx+1))*HID + c];
    }
  }
  out[i] = f2bf(gelu_f(acc));
}

// ---------------- local 7x7 window attention: one workgroup per window, all 4 heads
__global__ __launch_bounds__(256) void lattn_k(const float* __restrict__ qkv, u16* __restrict__ out){
  int blk = blockIdx.x;
  int wxv = blk & 15, wyv = (blk >> 4) & 15, b = blk >> 8;
  __shared__ __align__(16) u16 Qs [NHD][64*32];
  __shared__ __align__(16) u16 Ks [NHD][64*32];
  __shared__ __align__(16) u16 Vts[NHD][32*64];
  __shared__ __align__(16) u16 Ps [8][16*32];
  for (int idx = threadIdx.x; idx < 3*NHD*64*32; idx += 256){
    int d = idx & 31;
    int row = (idx >> 5) & 63;
    int h = (idx >> 11) & 3;
    int s = idx >> 13;
    u16 hv = 0;
    if (row < 49){
      int py = row / 7, px = row % 7;
      int gt = (wyv*7 + py)*WW2 + wxv*7 + px;
      hv = f2bf(qkv[((size_t)(b*NTOK + gt))*384 + s*128 + h*32 + d]);
    }
    if (s == 0)      Qs [h][row*32 + d] = hv;
    else if (s == 1) Ks [h][row*32 + d] = hv;
    else             Vts[h][d*64 + row] = hv;
  }
  __syncthreads();
  int wave = threadIdx.x >> 5, lane = threadIdx.x & 31;
  int l15 = lane & 15, hi = lane >> 4;
  const float scale = 0.1767766952966369f;  // 1/sqrt(32)
  for (int task = wave; task < 16; task += 8){
    int h = task >> 2, qt = task & 3;
    v16bf aq = load_frag(&Qs[h][(qt*16 + l15)*32], 0, lane);
    v8f acc[4];
#pragma unroll
    for (int nt=0;nt<4;++nt){
      acc[nt] = vzero();
      v16bf bk = load_frag(&Ks[h][(nt*16 + l15)*32], 0, lane);
      wmma(acc[nt], aq, bk);
    }
    float rmax[8], rsum[8];
#pragma unroll
    for (int r=0;r<8;++r) rmax[r] = -1e30f;
#pragma unroll
    for (int nt=0;nt<4;++nt){
      bool valid = (nt*16 + l15) < 49;
#pragma unroll
      for (int r=0;r<8;++r){
        float vv = valid ? acc[nt][r]*scale : -1e30f;
        acc[nt][r] = vv;
        rmax[r] = fmaxf(rmax[r], vv);
      }
    }
#pragma unroll
    for (int r=0;r<8;++r){
#pragma unroll
      for (int m=1;m<16;m<<=1) rmax[r] = fmaxf(rmax[r], __shfl_xor(rmax[r], m, 32));
      rsum[r] = 0.f;
    }
#pragma unroll
    for (int nt=0;nt<4;++nt){
#pragma unroll
      for (int r=0;r<8;++r){ float e = __expf(acc[nt][r]-rmax[r]); acc[nt][r]=e; rsum[r]+=e; }
    }
#pragma unroll
    for (int r=0;r<8;++r){
#pragma unroll
      for (int m=1;m<16;m<<=1) rsum[r] += __shfl_xor(rsum[r], m, 32);
      rsum[r] = 1.f / rsum[r];
    }
    v8f o0 = vzero(), o1 = vzero();
#pragma unroll
    for (int ks=0;ks<2;++ks){
#pragma unroll
      for (int j=0;j<2;++j){
        int nt = 2*ks + j;
#pragma unroll
        for (int r=0;r<8;++r)
          Ps[wave][(r + 8*hi)*32 + j*16 + l15] = f2bf(acc[nt][r]*rsum[r]);
      }
      v16bf apf = load_frag(&Ps[wave][l15*32], 0, lane);
      v16bf bv0 = load_frag(&Vts[h][l15*64],      ks*32, lane);
      v16bf bv1 = load_frag(&Vts[h][(16+l15)*64], ks*32, lane);
      wmma(o0, apf, bv0);
      wmma(o1, apf, bv1);
    }
#pragma unroll
    for (int r=0;r<8;++r){
      int row = qt*16 + r + 8*hi;
      if (row < 49){
        int py = row/7, px = row%7;
        int gt = (wyv*7+py)*WW2 + wxv*7+px;
        size_t base = ((size_t)(b*NTOK + gt))*CC + h*HD;
        out[base + l15]      = f2bf(o0[r]);
        out[base + 16 + l15] = f2bf(o1[r]);
      }
    }
  }
}

// ---------------- global SR attention: one workgroup per (b, head, 128-query block)
__global__ __launch_bounds__(256) void gattn_k(const u16* __restrict__ q, const u16* __restrict__ kv,
                                               u16* __restrict__ out){
  int blk = blockIdx.x;
  int qb = blk % 98; int h = (blk/98) & 3; int b = blk / 392;
  __shared__ __align__(16) u16 Ksh [MKVP*32];
  __shared__ __align__(16) u16 Vtsh[32*MKVP];
  __shared__ __align__(16) u16 Ps  [8][16*32];
  for (int idx = threadIdx.x; idx < 2*MKVP*32; idx += 256){
    int d = idx & 31;
    int m = (idx >> 5) % MKVP;
    int s = idx / (MKVP*32);
    u16 hv = 0;
    if (m < MKV) hv = kv[((size_t)(b*MKV + m))*256 + s*128 + h*32 + d];
    if (s == 0) Ksh[m*32 + d] = hv; else Vtsh[d*MKVP + m] = hv;
  }
  __syncthreads();
  int wave = threadIdx.x >> 5, lane = threadIdx.x & 31;
  int l15 = lane & 15, hi = lane >> 4;
  int qrow = qb*128 + wave*16 + l15;
  v16bf aq = load_frag(q + ((size_t)(b*NTOK + qrow))*CC + h*HD, 0, lane);
  v8f acc[14];
#pragma unroll
  for (int nt=0;nt<14;++nt){
    acc[nt] = vzero();
    v16bf bk = load_frag(&Ksh[(nt*16 + l15)*32], 0, lane);
    wmma(acc[nt], aq, bk);
  }
  const float scale = 0.1767766952966369f;
  float rmax[8], rsum[8];
#pragma unroll
  for (int r=0;r<8;++r) rmax[r] = -1e30f;
#pragma unroll
  for (int nt=0;nt<14;++nt){
    bool valid = (nt*16 + l15) < MKV;
#pragma unroll
    for (int r=0;r<8;++r){
      float vv = valid ? acc[nt][r]*scale : -1e30f;
      acc[nt][r] = vv;
      rmax[r] = fmaxf(rmax[r], vv);
    }
  }
#pragma unroll
  for (int r=0;r<8;++r){
#pragma unroll
    for (int m=1;m<16;m<<=1) rmax[r] = fmaxf(rmax[r], __shfl_xor(rmax[r], m, 32));
    rsum[r] = 0.f;
  }
#pragma unroll
  for (int nt=0;nt<14;++nt){
#pragma unroll
    for (int r=0;r<8;++r){ float e = __expf(acc[nt][r]-rmax[r]); acc[nt][r]=e; rsum[r]+=e; }
  }
#pragma unroll
  for (int r=0;r<8;++r){
#pragma unroll
    for (int m=1;m<16;m<<=1) rsum[r] += __shfl_xor(rsum[r], m, 32);
    rsum[r] = 1.f / rsum[r];
  }
  v8f o0 = vzero(), o1 = vzero();
#pragma unroll
  for (int ks=0;ks<7;++ks){
#pragma unroll
    for (int j=0;j<2;++j){
      int nt = 2*ks + j;
#pragma unroll
      for (int r=0;r<8;++r)
        Ps[wave][(r + 8*hi)*32 + j*16 + l15] = f2bf(acc[nt][r]*rsum[r]);
    }
    v16bf apf = load_frag(&Ps[wave][l15*32], 0, lane);
    v16bf bv0 = load_frag(&Vtsh[l15*MKVP],      ks*32, lane);
    v16bf bv1 = load_frag(&Vtsh[(16+l15)*MKVP], ks*32, lane);
    wmma(o0, apf, bv0);
    wmma(o1, apf, bv1);
  }
#pragma unroll
  for (int r=0;r<8;++r){
    int row = qb*128 + wave*16 + r + 8*hi;
    size_t base = ((size_t)(b*NTOK + row))*CC + h*HD;
    out[base + l15]      = f2bf(o0[r]);
    out[base + 16 + l15] = f2bf(o1[r]);
  }
}

// ---------------- host orchestration ----------------
extern "C" void kernel_launch(void* const* d_in, const int* in_sizes, int n_in,
                              void* d_out, int out_size, void* d_ws, size_t ws_size,
                              hipStream_t stream){
  (void)in_sizes; (void)n_in; (void)out_size; (void)ws_size;
  const float* x0   = (const float*)d_in[0];
  const float* ln1g = (const float*)d_in[1];  const float* ln1b = (const float*)d_in[2];
  const float* qkvw = (const float*)d_in[3];  const float* qkvb = (const float*)d_in[4];
  const float* lpw  = (const float*)d_in[5];  const float* lpb  = (const float*)d_in[6];
  const float* ln2g = (const float*)d_in[7];  const float* ln2b = (const float*)d_in[8];
  const float* lf1w = (const float*)d_in[9];  const float* lf1b = (const float*)d_in[10];
  const float* ldww = (const float*)d_in[11]; const float* ldwb = (const float*)d_in[12];
  const float* lf2w = (const float*)d_in[13]; const float* lf2b = (const float*)d_in[14];
  const float* gn1g = (const float*)d_in[15]; const float* gn1b = (const float*)d_in[16];
  const float* gqw  = (const float*)d_in[17]; const float* gqb  = (const float*)d_in[18];
  const float* gkvw = (const float*)d_in[19]; const float* gkvb = (const float*)d_in[20];
  const float* gsrw = (const float*)d_in[21]; const float* gsrb = (const float*)d_in[22];
  const float* gsng = (const float*)d_in[23]; const float* gsnb = (const float*)d_in[24];
  const float* gpw  = (const float*)d_in[25]; const float* gpb  = (const float*)d_in[26];
  const float* gn2g = (const float*)d_in[27]; const float* gn2b = (const float*)d_in[28];
  const float* gf1w = (const float*)d_in[29]; const float* gf1b = (const float*)d_in[30];
  const float* gdww = (const float*)d_in[31]; const float* gdwb = (const float*)d_in[32];
  const float* gf2w = (const float*)d_in[33]; const float* gf2b = (const float*)d_in[34];
  float* outp = (float*)d_out;

  char* wsp = (char*)d_ws; size_t off = 0;
  auto alloc = [&](size_t bytes)->void*{ void* p = wsp + off; off += (bytes + 255) & ~(size_t)255; return p; };
  u16* wt_qkv = (u16*)alloc((size_t)384*128*2);
  u16* wt_pw  = (u16*)alloc((size_t)128*128*2);
  u16* wt_f1  = (u16*)alloc((size_t)512*128*2);
  u16* wt_f2  = (u16*)alloc((size_t)128*512*2);
  u16* wt_gq  = (u16*)alloc((size_t)128*128*2);
  u16* wt_gkv = (u16*)alloc((size_t)256*128*2);
  u16* wt_gsr = (u16*)alloc((size_t)128*8192*2);
  u16* wt_gpw = (u16*)alloc((size_t)128*128*2);
  u16* wt_gf1 = (u16*)alloc((size_t)512*128*2);
  u16* wt_gf2 = (u16*)alloc((size_t)128*512*2);
  u16*   lnb  = (u16*)  alloc((size_t)BNTOK*CC*2);
  float* bigf = (float*)alloc((size_t)BNTOK*HID*4);   // qkv / mlp hidden fp32
  u16*   bigh = (u16*)  alloc((size_t)BNTOK*HID*2);   // attn out / gelu hidden bf16
  u16*   qb16 = (u16*)  alloc((size_t)BNTOK*CC*2);
  u16*   asr  = (u16*)  alloc((size_t)1568*8192*2);
  float* xsf  = (float*)alloc((size_t)1568*CC*4);
  u16*   lns  = (u16*)  alloc((size_t)1568*CC*2);
  u16*   kvb  = (u16*)  alloc((size_t)1568*256*2);
  float* Xa   = (float*)alloc((size_t)BNTOK*CC*4);
  float* Xb   = (float*)alloc((size_t)BNTOK*CC*4);

  auto wt = [&](const float* W, u16* Wt, int K, int N){
    int n = K*N; wtr_k<<<(n+255)/256,256,0,stream>>>(W,Wt,K,N);
  };
  auto gemm  = [&](const u16* A, const u16* Wt, const float* bias, int M, int N, int K, float* oF){
    dim3 g((M+127)/128, N/64);
    gemm_k<false,false><<<g,256,0,stream>>>(A,Wt,bias,nullptr,oF,nullptr,M,N,K);
  };
  auto gemmR = [&](const u16* A, const u16* Wt, const float* bias, const float* res, int M, int N, int K, float* oF){
    dim3 g((M+127)/128, N/64);
    gemm_k<true,false><<<g,256,0,stream>>>(A,Wt,bias,res,oF,nullptr,M,N,K);
  };
  auto gemmB = [&](const u16* A, const u16* Wt, const float* bias, int M, int N, int K, u16* oB){
    dim3 g((M+127)/128, N/64);
    gemm_k<false,true><<<g,256,0,stream>>>(A,Wt,bias,nullptr,nullptr,oB,M,N,K);
  };

  // weight prep (bf16, transposed to [N,K])
  wt(qkvw, wt_qkv, 128, 384);
  wt(lpw , wt_pw , 128, 128);
  wt(lf1w, wt_f1 , 128, 512);
  wt(lf2w, wt_f2 , 512, 128);
  wt(gqw , wt_gq , 128, 128);
  wt(gkvw, wt_gkv, 128, 256);
  wt(gsrw, wt_gsr, 8192, 128);
  wt(gpw , wt_gpw, 128, 128);
  wt(gf1w, wt_gf1, 128, 512);
  wt(gf2w, wt_gf2, 512, 128);

  // ---- local attention block ----
  ln_k<<<BNTOK/8,256,0,stream>>>(x0, ln1g, ln1b, lnb, BNTOK);
  gemm(lnb, wt_qkv, qkvb, BNTOK, 384, 128, bigf);                 // qkv fp32
  lattn_k<<<BATCH*256,256,0,stream>>>(bigf, bigh);                // window attn -> bf16
  gemmR(bigh, wt_pw, lpb, x0, BNTOK, 128, 128, Xa);               // proj + residual
  ln_k<<<BNTOK/8,256,0,stream>>>(Xa, ln2g, ln2b, lnb, BNTOK);
  gemm(lnb, wt_f1, lf1b, BNTOK, 512, 128, bigf);                  // fc1 fp32
  dw_k<<<(BNTOK*HID)/256,256,0,stream>>>(bigf, ldww, ldwb, bigh, BNTOK*HID); // dwconv+gelu -> bf16
  gemmR(bigh, wt_f2, lf2b, Xa, BNTOK, 128, 512, Xb);              // fc2 + residual

  // ---- global SR attention block ----
  ln_k<<<BNTOK/8,256,0,stream>>>(Xb, gn1g, gn1b, lnb, BNTOK);
  gemmB(lnb, wt_gq, gqb, BNTOK, 128, 128, qb16);                  // q bf16
  im2col_k<<<(1568*8192)/256,256,0,stream>>>(lnb, asr);           // im2col for SR conv
  gemm(asr, wt_gsr, gsrb, 1568, 128, 8192, xsf);                  // SR conv as GEMM
  ln_k<<<(1568+7)/8,256,0,stream>>>(xsf, gsng, gsnb, lns, 1568);  // sr-LN
  gemmB(lns, wt_gkv, gkvb, 1568, 256, 128, kvb);                  // kv bf16
  gattn_k<<<BATCH*4*98,256,0,stream>>>(qb16, kvb, bigh);          // global attn -> bf16
  gemmR(bigh, wt_gpw, gpb, Xb, BNTOK, 128, 128, Xa);              // proj + residual
  ln_k<<<BNTOK/8,256,0,stream>>>(Xa, gn2g, gn2b, lnb, BNTOK);
  gemm(lnb, wt_gf1, gf1b, BNTOK, 512, 128, bigf);                 // fc1
  dw_k<<<(BNTOK*HID)/256,256,0,stream>>>(bigf, gdww, gdwb, bigh, BNTOK*HID);
  gemmR(bigh, wt_gf2, gf2b, Xa, BNTOK, 128, 512, outp);           // fc2 + residual -> out
}
